// dis_MaskedDirectMultiheadAttention_11115375362511
// MI455X (gfx1250) — compile-verified, matrix operand order checked by reading, not run on hardware
//
#include <hip/hip_runtime.h>
#include <hip/hip_bf16.h>
#include <stdint.h>

// ---------------------------------------------------------------------------
// MaskedDirectMultiheadAttention for MI455X (gfx1250, wave32, WMMA bf16)
// B=4, L=2048, D=256.  out = softmax(masked(atten)) @ (value@Wv+bv) @ Wo + bo
// Heads share the attention matrix -> collapses to per-batch (LxL)@(LxD).
// Single-pass (flash-style) online softmax: atten/mask/pad read exactly once,
// with explicit software pipelining of the streaming loads.
// ---------------------------------------------------------------------------

#define BB 4
#define LL 2048
#define DD 256

typedef __attribute__((ext_vector_type(16))) __bf16 v16bf;
typedef __attribute__((ext_vector_type(8)))  float  v8f;

union FragB { v16bf v; uint4 q[2]; };

static __device__ __forceinline__ unsigned short bfbits(float f) {
    __bf16 h = (__bf16)f;
    return __builtin_bit_cast(unsigned short, h);
}

static __device__ __forceinline__ v8f wmma_bf16(v16bf a, v16bf b, v8f c) {
    // (neg_a, A, neg_b, B, c_mod, C, reuse_a, reuse_b)
    return __builtin_amdgcn_wmma_f32_16x16x32_bf16(false, a, false, b, (short)0, c, false, false);
}

// ---------------------------------------------------------------------------
// Prep kernels
// ---------------------------------------------------------------------------
__global__ void cvt_bf16_kernel(const float* __restrict__ in,
                                unsigned short* __restrict__ out, int n) {
    int i = blockIdx.x * blockDim.x + threadIdx.x;
    if (i < n) out[i] = bfbits(in[i]);
}

// WT[n*256 + d] = bf16(W[d*256 + n])   (256x256)
__global__ void transpose_bf16_kernel(const float* __restrict__ W,
                                      unsigned short* __restrict__ WT) {
    int d = blockIdx.x;
    int n = threadIdx.x;
    WT[n * DD + d] = bfbits(W[d * DD + n]);
}

// ---------------------------------------------------------------------------
// Stage 1: v = value @ Wv + bv, stored TRANSPOSED bf16: vT[(b*256+n)*2048 + j]
// Block = 256 thr (8 waves); block owns 16 rows x 256 cols; wave w does col
// tiles n=w*16 and n=w*16+128.  K = 256, step 32.
// ---------------------------------------------------------------------------
__global__ void vproj_kernel(const unsigned short* __restrict__ valBf,
                             const unsigned short* __restrict__ WvT,
                             const float* __restrict__ bv,
                             unsigned short* __restrict__ vT) {
    int wave = threadIdx.x >> 5;
    int lane = threadIdx.x & 31;
    int R0 = blockIdx.x * 16;                // global row base (0..8191)
    int b  = R0 >> 11;
    int j0 = R0 & (LL - 1);
    int laneN = lane & 15;
    int kb = (lane >> 4) * 8;                // upper half-wave K offset
    int n0 = wave * 16;
    int n1 = n0 + 128;

    v8f acc0 = {}; v8f acc1 = {};
    for (int kk = 0; kk < DD; kk += 32) {
        FragB a, b0, b1;
        const uint4* ap  = (const uint4*)(valBf + (size_t)(R0 + laneN) * DD + kk + kb);
        a.q[0] = ap[0];  a.q[1] = ap[2];
        const uint4* bp0 = (const uint4*)(WvT + (size_t)(n0 + laneN) * DD + kk + kb);
        b0.q[0] = bp0[0]; b0.q[1] = bp0[2];
        const uint4* bp1 = (const uint4*)(WvT + (size_t)(n1 + laneN) * DD + kk + kb);
        b1.q[0] = bp1[0]; b1.q[1] = bp1[2];
        acc0 = wmma_bf16(a.v, b0.v, acc0);
        acc1 = wmma_bf16(a.v, b1.v, acc1);
    }
    // store transposed: lane holds col n, 8 consecutive rows -> one b128 store
    {
        int n = n0 + laneN;
        float bvn = bv[n];
        uint4 u;
        u.x = (unsigned)bfbits(acc0[0] + bvn) | ((unsigned)bfbits(acc0[1] + bvn) << 16);
        u.y = (unsigned)bfbits(acc0[2] + bvn) | ((unsigned)bfbits(acc0[3] + bvn) << 16);
        u.z = (unsigned)bfbits(acc0[4] + bvn) | ((unsigned)bfbits(acc0[5] + bvn) << 16);
        u.w = (unsigned)bfbits(acc0[6] + bvn) | ((unsigned)bfbits(acc0[7] + bvn) << 16);
        *(uint4*)(vT + ((size_t)(b * DD + n)) * LL + j0 + kb) = u;
    }
    {
        int n = n1 + laneN;
        float bvn = bv[n];
        uint4 u;
        u.x = (unsigned)bfbits(acc1[0] + bvn) | ((unsigned)bfbits(acc1[1] + bvn) << 16);
        u.y = (unsigned)bfbits(acc1[2] + bvn) | ((unsigned)bfbits(acc1[3] + bvn) << 16);
        u.z = (unsigned)bfbits(acc1[4] + bvn) | ((unsigned)bfbits(acc1[5] + bvn) << 16);
        u.w = (unsigned)bfbits(acc1[6] + bvn) | ((unsigned)bfbits(acc1[7] + bvn) << 16);
        *(uint4*)(vT + ((size_t)(b * DD + n)) * LL + j0 + kb) = u;
    }
}

// ---------------------------------------------------------------------------
// Stage 2 (dominant): O1 = softmax(masked(atten)) @ v, per batch, single pass.
// Block = 256 thr (8 waves) owns a 16-row x 256-col output tile.
// Online softmax; per 32-wide K-step:
//   issue vT B-frag loads (only depend on kk)      } overlap with
//   prefetch next step's atten/mask/pad to regs    } everything below
//   producers: masked x; width-16 shfl max; p=exp(x-m_new) -> bf16 LDS tile;
//              publish scale=exp(m_old-m_new) in LDS; update running sum
//   barrier;  all lanes: acc *= scale[row]; 2x WMMA;  barrier
// Epilogue: acc *= 1/l.
// ---------------------------------------------------------------------------
__global__ void attn_pv_kernel(const float* __restrict__ atten,
                               const float* __restrict__ mask,
                               const int*   __restrict__ pad,
                               const unsigned short* __restrict__ vT,
                               unsigned short* __restrict__ O1) {
    __shared__ __align__(16) unsigned short ptile[16 * 32];
    __shared__ float sc[16];

    int t    = threadIdx.x;
    int wave = t >> 5;
    int lane = t & 31;
    int R0 = blockIdx.x * 16;                // global row base
    int b  = R0 >> 11;

    // P-producer role: thread t fills row prow, cols pc0..pc0+1
    int prow = t >> 4;                       // 0..15 (constant per thread)
    int pc0  = (t & 15) * 2;
    size_t abase = (size_t)(R0 + prow) * LL + pc0;   // even -> float2 aligned
    const float2* a2 = (const float2*)(atten + abase);
    const float2* m2 = (const float2*)(mask + abase);
    const int2*   p2 = (const int2*)(pad + abase);

    float mrun = -INFINITY;                  // running row max
    float lrun = 0.f;                        // running row sum

    // WMMA role
    int laneN = lane & 15;
    int kb = (lane >> 4) * 8;
    int n0 = wave * 32;
    int n1 = n0 + 16;
    const unsigned short* vb0 = vT + ((size_t)(b * DD + n0 + laneN)) * LL + kb;
    const unsigned short* vb1 = vT + ((size_t)(b * DD + n1 + laneN)) * LL + kb;

    // prime the producer pipeline (step kk=0)
    float2 av = a2[0];
    float2 mv = m2[0];
    int2   pv = p2[0];

    v8f acc0 = {}; v8f acc1 = {};
    for (int kk = 0; kk < LL; kk += 32) {
        // ---- issue B-frag loads early (independent of LDS/barriers) ----
        FragB b0, b1;
        const uint4* bp0 = (const uint4*)(vb0 + kk);
        b0.q[0] = bp0[0]; b0.q[1] = bp0[2];
        const uint4* bp1 = (const uint4*)(vb1 + kk);
        b1.q[0] = bp1[0]; b1.q[1] = bp1[2];

        // ---- prefetch next step's streaming data ----
        float2 av_n = {0.f, 0.f};
        float2 mv_n = {0.f, 0.f};
        int2   pv_n = {0, 0};
        if (kk + 32 < LL) {
            av_n = a2[(kk + 32) >> 1];
            mv_n = m2[(kk + 32) >> 1];
            pv_n = p2[(kk + 32) >> 1];
        }

        // ---- producer: masked logits for 2 columns (current step) ----
        float x0 = (pv.x == 0) ? -INFINITY : ((mv.x < 0.5f) ? -3.4028235e38f : av.x);
        float x1 = (pv.y == 0) ? -INFINITY : ((mv.y < 0.5f) ? -3.4028235e38f : av.y);

        // row max over this 32-wide step (16 lanes, width-16 xor reduce)
        float mloc = fmaxf(x0, x1);
        mloc = fmaxf(mloc, __shfl_xor(mloc, 8, 16));
        mloc = fmaxf(mloc, __shfl_xor(mloc, 4, 16));
        mloc = fmaxf(mloc, __shfl_xor(mloc, 2, 16));
        mloc = fmaxf(mloc, __shfl_xor(mloc, 1, 16));

        float mnew = fmaxf(mrun, mloc);
        bool dead = (mnew == -INFINITY);     // entire row prefix is -inf
        float e0 = __expf(x0 - mnew);        // NaN only if dead (discarded)
        float e1 = __expf(x1 - mnew);
        float p0 = dead ? 0.f : e0;
        float p1 = dead ? 0.f : e1;
        float es = __expf(mrun - mnew);      // NaN only if dead (discarded)
        float scale = dead ? 1.f : ((mrun > -INFINITY) ? es : 0.f);

        // row sum of this step
        float ps = p0 + p1;
        ps += __shfl_xor(ps, 8, 16);
        ps += __shfl_xor(ps, 4, 16);
        ps += __shfl_xor(ps, 2, 16);
        ps += __shfl_xor(ps, 1, 16);
        lrun = lrun * scale + ps;
        mrun = mnew;

        unsigned int ppk = (unsigned)bfbits(p0) | ((unsigned)bfbits(p1) << 16);
        *(unsigned int*)(ptile + prow * 32 + pc0) = ppk;
        if ((t & 15) == 0) sc[prow] = scale;
        __syncthreads();

        // ---- consumer: rescale accumulators, then WMMA ----
#pragma unroll
        for (int r = 0; r < 8; ++r) {
            float s = sc[kb + r];
            acc0[r] *= s;
            acc1[r] *= s;
        }
        FragB a;
        const uint4* ap = (const uint4*)(ptile + (laneN * 32 + kb));
        a.q[0] = ap[0];  a.q[1] = ap[2];
        acc0 = wmma_bf16(a.v, b0.v, acc0);
        acc1 = wmma_bf16(a.v, b1.v, acc1);

        __syncthreads();

        // rotate pipeline registers
        av = av_n; mv = mv_n; pv = pv_n;
    }

    // ---- epilogue: normalize by 1/l and store O1 row-major bf16 ----
    if ((t & 15) == 0) sc[prow] = 1.f / lrun;
    __syncthreads();

    int m0 = R0 + kb;                        // 8 consecutive output rows
#pragma unroll
    for (int r = 0; r < 8; ++r) {
        float s = sc[kb + r];
        size_t o = (size_t)(m0 + r) * DD;
        O1[o + n0 + laneN] = bfbits(acc0[r] * s);
        O1[o + n1 + laneN] = bfbits(acc1[r] * s);
    }
}

// ---------------------------------------------------------------------------
// Stage 3: out = O1 @ Wo + bo  (f32 output)
// ---------------------------------------------------------------------------
__global__ void oproj_kernel(const unsigned short* __restrict__ O1,
                             const unsigned short* __restrict__ WoT,
                             const float* __restrict__ bo,
                             float* __restrict__ out) {
    int wave = threadIdx.x >> 5;
    int lane = threadIdx.x & 31;
    int R0 = blockIdx.x * 16;
    int laneN = lane & 15;
    int kb = (lane >> 4) * 8;
    int n0 = wave * 16;
    int n1 = n0 + 128;

    v8f acc0 = {}; v8f acc1 = {};
    for (int kk = 0; kk < DD; kk += 32) {
        FragB a, b0, b1;
        const uint4* ap  = (const uint4*)(O1 + (size_t)(R0 + laneN) * DD + kk + kb);
        a.q[0] = ap[0];  a.q[1] = ap[2];
        const uint4* bp0 = (const uint4*)(WoT + (size_t)(n0 + laneN) * DD + kk + kb);
        b0.q[0] = bp0[0]; b0.q[1] = bp0[2];
        const uint4* bp1 = (const uint4*)(WoT + (size_t)(n1 + laneN) * DD + kk + kb);
        b1.q[0] = bp1[0]; b1.q[1] = bp1[2];
        acc0 = wmma_bf16(a.v, b0.v, acc0);
        acc1 = wmma_bf16(a.v, b1.v, acc1);
    }
    int m0 = R0 + kb;
    float bn0 = bo[n0 + laneN];
    float bn1 = bo[n1 + laneN];
#pragma unroll
    for (int r = 0; r < 8; ++r) {
        size_t o = (size_t)(m0 + r) * DD;
        out[o + n0 + laneN] = acc0[r] + bn0;
        out[o + n1 + laneN] = acc1[r] + bn1;
    }
}

// ---------------------------------------------------------------------------
// Launch
// ---------------------------------------------------------------------------
extern "C" void kernel_launch(void* const* d_in, const int* in_sizes, int n_in,
                              void* d_out, int out_size, void* d_ws, size_t ws_size,
                              hipStream_t stream) {
    (void)in_sizes; (void)n_in; (void)out_size; (void)ws_size;

    const float* atten = (const float*)d_in[0];  // (B,L,L)
    const float* value = (const float*)d_in[1];  // (B,L,D)
    const float* mask  = (const float*)d_in[2];  // (B,1,L,L)
    const int*   pad   = (const int*)d_in[3];    // (B,L,L)
    const float* Wv    = (const float*)d_in[4];  // (D,D)
    const float* bv    = (const float*)d_in[5];  // (D,)
    const float* Wo    = (const float*)d_in[6];  // (D,D)
    const float* bo    = (const float*)d_in[7];  // (D,)
    float* out = (float*)d_out;                  // (B,1,L,D) f32

    // workspace layout (bytes)
    char* ws = (char*)d_ws;
    const size_t szValBf = (size_t)BB * LL * DD * 2;   // 4 MB
    const size_t szVT    = (size_t)BB * DD * LL * 2;   // 4 MB
    const size_t szO1    = (size_t)BB * LL * DD * 2;   // 4 MB
    const size_t szWT    = (size_t)DD * DD * 2;        // 128 KB

    unsigned short* valueBf = (unsigned short*)ws;            ws += szValBf;
    unsigned short* vT      = (unsigned short*)ws;            ws += szVT;
    unsigned short* O1      = (unsigned short*)ws;            ws += szO1;
    unsigned short* WvT     = (unsigned short*)ws;            ws += szWT;
    unsigned short* WoT     = (unsigned short*)ws;            ws += szWT;

    const int nVal = BB * LL * DD;  // 2,097,152
    cvt_bf16_kernel<<<(nVal + 255) / 256, 256, 0, stream>>>(value, valueBf, nVal);
    transpose_bf16_kernel<<<DD, DD, 0, stream>>>(Wv, WvT);
    transpose_bf16_kernel<<<DD, DD, 0, stream>>>(Wo, WoT);

    vproj_kernel<<<(BB * LL) / 16, 256, 0, stream>>>(valueBf, WvT, bv, vT);

    attn_pv_kernel<<<(BB * LL) / 16, 256, 0, stream>>>(atten, mask, pad, vT, O1);

    oproj_kernel<<<(BB * LL) / 16, 256, 0, stream>>>(O1, WoT, bo, out);
}
